// ComplexLSTMCell_6949257085132
// MI455X (gfx1250) — matrix operand order repaired and checked by reading, not compiled, not run on hardware
//
#include <hip/hip_runtime.h>

// ---------------- Types for WMMA ----------------
typedef __attribute__((ext_vector_type(16))) __bf16 v16bf;
typedef __attribute__((ext_vector_type(8)))  float  v8f;

union FragU {
    v16bf v;
    uint4 q[2];
};

// fp32 -> bf16 (round-to-nearest-even-ish), stored as ushort
__device__ __forceinline__ unsigned short f2bf(float f) {
    union { float f; unsigned int u; } x; x.f = f;
    unsigned int u = x.u;
    unsigned int r = u + 0x7FFFu + ((u >> 16) & 1u);
    return (unsigned short)(r >> 16);
}

__device__ __forceinline__ float sigm_f(float x) {
    return 1.0f / (1.0f + __expf(-x));
}
__device__ __forceinline__ float tanh_f(float x) {
    return 1.0f - 2.0f / (__expf(2.0f * x) + 1.0f);
}

// Problem constants
#define BATCH 256
#define TT    512
#define KTOT  384      // 128 (xr|xi) + 256 (hr|hi)
#define NTOT  1024     // 512 zr + 512 zi
#define BROWS 16       // batch rows per workgroup
#define AS    392      // padded LDS stride for A (multiple of 8 elems -> 16B-aligned rows)

#define WMMA_BF16(A, B, C) \
    __builtin_amdgcn_wmma_f32_16x16x32_bf16(false, (A), false, (B), (short)0, (C), false, false)

// ---------------- Prep: pack combined transposed weights Wt[n][k] in bf16 ----------------
// zr (n<512):  k<64: Kr[k][n]       64<=k<128: -Ki[k-64][n]
//              128<=k<256: Rr[k-128][n]   256<=k<384: -Ri[k-256][n]
// zi (n>=512): k<64: Ki[k][n']      64<=k<128:  Kr[k-64][n']
//              128<=k<256: Ri[k-128][n']  256<=k<384:  Rr[k-256][n']
__global__ __launch_bounds__(256) void pack_weights_kernel(
    const float* __restrict__ Kr, const float* __restrict__ Ki,
    const float* __restrict__ Rr, const float* __restrict__ Ri,
    unsigned short* __restrict__ Wt)
{
    int idx = blockIdx.x * 256 + threadIdx.x;
    if (idx >= NTOT * KTOT) return;
    int n = idx / KTOT;
    int k = idx - n * KTOT;
    int im = (n >= 512);
    int np = im ? (n - 512) : n;
    float v;
    if (k < 64) {
        v = im ? Ki[k * 512 + np] : Kr[k * 512 + np];
    } else if (k < 128) {
        int c = k - 64;
        v = im ? Kr[c * 512 + np] : -Ki[c * 512 + np];
    } else if (k < 256) {
        int c = k - 128;
        v = im ? Ri[c * 512 + np] : Rr[c * 512 + np];
    } else {
        int c = k - 256;
        v = im ? Rr[c * 512 + np] : -Ri[c * 512 + np];
    }
    Wt[idx] = f2bf(v);
}

// ---------------- Persistent recurrent kernel: 16 WGs x (16 batch rows each) ----------------
__global__ __launch_bounds__(256) void complex_lstm_kernel(
    const float* __restrict__ x,        // [256,512,128]
    const float* __restrict__ h0,       // [256,256]
    const float* __restrict__ c0,       // [256,256]
    const unsigned short* __restrict__ Wt, // [1024][384] bf16, N-major
    const float* __restrict__ bias_r,   // [512]
    const float* __restrict__ bias_i,   // [512]
    float* __restrict__ out)            // [256,512,256]
{
    __shared__ __align__(16) unsigned short Abuf[BROWS * AS];   // [16][392] bf16: x(0..127) | h(128..383)
    __shared__ __align__(16) float          Zbuf[BROWS * NTOT]; // [16][1024] fp32 pre-activations
    __shared__ __align__(16) float          Cbuf[BROWS * 256];  // [16][256] fp32 cell state (cr|ci)

    const int tid  = threadIdx.x;
    const int wg   = blockIdx.x;       // 0..15
    const int m0   = wg * BROWS;       // batch base
    const int lane = tid & 31;
    const int wave = tid >> 5;         // 0..7, each owns 128 N columns

    // ---- stage x_t into Abuf[:, 0..127] (16 rows x 128 cols, bf16) ----
    auto stageX = [&](int t) {
        int r = tid >> 4;            // 0..15
        int c = (tid & 15) * 8;      // 0..120
        const float* xp = x + (((size_t)(m0 + r) * TT + t) * 128 + c);
        float4 a = *(const float4*)xp;
        float4 b = *(const float4*)(xp + 4);
        unsigned short* dst = &Abuf[r * AS + c];
        dst[0] = f2bf(a.x); dst[1] = f2bf(a.y); dst[2] = f2bf(a.z); dst[3] = f2bf(a.w);
        dst[4] = f2bf(b.x); dst[5] = f2bf(b.y); dst[6] = f2bf(b.z); dst[7] = f2bf(b.w);
    };

    // ---- init: h0 -> Abuf (bf16), c0 -> Cbuf (fp32); stage x_0 ----
    for (int i = tid; i < BROWS * 256; i += 256) {
        int r = i >> 8;
        int u = i & 255;
        Abuf[r * AS + 128 + u] = f2bf(h0[(m0 + r) * 256 + u]);
        Cbuf[r * 256 + u]      = c0[(m0 + r) * 256 + u];
    }
    stageX(0);
    __syncthreads();

    // per-lane fragment geometry (CDNA5 wave32 WMMA layouts)
    const int am  = lane & 15;                 // A row (M)
    const int akh = (lane >> 4) * 8;           // A: K base per lane-half (+16 for hi block)
    const int bn  = wave * 128 + (lane & 15);  // B row in Wt (N column)
    const int bkh = (lane >> 4) * 8;           // B: K base per lane-half (+16 for hi block)

    // 32-bit "zero" offset laundered each step: keeps Wt's global address space
    // (global_load_b128, SGPR base) while preventing the (t-invariant) weight
    // fragment loads from being hoisted out of the recurrent loop.
    unsigned int zofs = 0;

    for (int t = 0; t < TT; ++t) {
        asm volatile("" : "+s"(zofs));
        const unsigned int bofs = zofs + (unsigned int)(bn * KTOT + bkh); // elements

        v8f acc[8];
        FragU a0, a1, b0[8], b1[8];

        auto loadA = [&](int kc, FragU& f) {
            const unsigned short* ap = &Abuf[am * AS + kc * 32 + akh];
            f.q[0] = *(const uint4*)(ap);
            f.q[1] = *(const uint4*)(ap + 16);
        };
        auto loadB = [&](int kc, FragU* f) {
            #pragma unroll
            for (int nt = 0; nt < 8; ++nt) {
                const unsigned short* bp = Wt + (bofs + (unsigned int)(nt * 16 * KTOT + kc * 32));
                f[nt].q[0] = *(const uint4*)(bp);
                f[nt].q[1] = *(const uint4*)(bp + 16);
            }
        };
        auto mmas = [&](FragU& a, FragU* b) {
            #pragma unroll
            for (int nt = 0; nt < 8; ++nt)
                acc[nt] = WMMA_BF16(a.v, b[nt].v, acc[nt]);
        };

        // ---- GEMM: [16 x 384] (LDS bf16) x [384 x 1024] (L2 bf16), double buffered ----
        // Peeled chunk 0: accumulator starts as inline 0 in the WMMA (no v_mov splat).
        loadA(0, a0);
        loadB(0, b0);
        loadA(1, a1);
        loadB(1, b1);
        {
            const v8f zero = {};
            #pragma unroll
            for (int nt = 0; nt < 8; ++nt)
                acc[nt] = WMMA_BF16(a0.v, b0[nt].v, zero);     // chunk 0, C = inline 0
        }
        loadA(2, a0);
        loadB(2, b0);
        mmas(a1, b1);                                          // chunk 1
        #pragma unroll 1
        for (int kc2 = 1; kc2 < 6; ++kc2) {
            int kcE = 2 * kc2;            // even chunk (in buf0)
            loadA(kcE + 1, a1);           // prefetch odd chunk
            loadB(kcE + 1, b1);
            mmas(a0, b0);                 // compute even
            if (kc2 < 5) {                // prefetch next even chunk
                loadA(kcE + 2, a0);
                loadB(kcE + 2, b0);
            }
            mmas(a1, b1);                 // compute odd
        }

        // scatter z tiles to LDS (C/D layout: lanes 0-15 -> M=j, lanes 16-31 -> M=8+j, N=lane&15)
        {
            int mb = (lane >> 4) * 8;
            #pragma unroll
            for (int nt = 0; nt < 8; ++nt) {
                int n = wave * 128 + nt * 16 + (lane & 15);
                #pragma unroll
                for (int j = 0; j < 8; ++j)
                    Zbuf[(mb + j) * NTOT + n] = acc[nt][j];
            }
        }
        __syncthreads();

        // ---- stage next x (overlaps with elementwise transcendentals) ----
        if (t + 1 < TT) stageX(t + 1);

        // ---- fused complex-LSTM elementwise update: 16 rows x 128 units ----
        for (int i = tid; i < BROWS * 128; i += 256) {
            int r = i >> 7;
            int u = i & 127;
            const float* zrow = &Zbuf[r * NTOT];
            float z0r = zrow[      u] + bias_r[      u];
            float z1r = zrow[128 + u] + bias_r[128 + u];
            float z2r = zrow[256 + u] + bias_r[256 + u];
            float z3r = zrow[384 + u] + bias_r[384 + u];
            float z0i = zrow[512 + u] + bias_i[      u];
            float z1i = zrow[640 + u] + bias_i[128 + u];
            float z2i = zrow[768 + u] + bias_i[256 + u];
            float z3i = zrow[896 + u] + bias_i[384 + u];

            float ir = sigm_f(z0r), fr = sigm_f(z1r);
            float ii = sigm_f(z0i), fi = sigm_f(z1i);
            float cr = Cbuf[r * 256 + u];
            float ci = Cbuf[r * 256 + 128 + u];
            float t2r = tanh_f(z2r), t2i = tanh_f(z2i);
            float crn = fr * cr - fi * ci + (ir * t2r - ii * t2i);
            float cin = fr * ci + fi * cr + (ir * t2i - ii * t2r);
            float orr = sigm_f(z3r), oi = sigm_f(z3i);
            float thr = tanh_f(crn), thi = tanh_f(cin);
            float hrn = orr * thr - oi * thi;
            float hin = orr * thi - oi * thr;

            Cbuf[r * 256 + u]       = crn;
            Cbuf[r * 256 + 128 + u] = cin;
            Abuf[r * AS + 128 + u]  = f2bf(hrn);   // next-step recurrent input
            Abuf[r * AS + 256 + u]  = f2bf(hin);

            size_t ob = ((size_t)(m0 + r) * TT + t) * 256;
            out[ob + u]       = hrn;
            out[ob + 128 + u] = hin;
        }
        __syncthreads();
    }
}

extern "C" void kernel_launch(void* const* d_in, const int* in_sizes, int n_in,
                              void* d_out, int out_size, void* d_ws, size_t ws_size,
                              hipStream_t stream) {
    const float* x   = (const float*)d_in[0];
    const float* h0  = (const float*)d_in[1];
    const float* c0  = (const float*)d_in[2];
    const float* Kr  = (const float*)d_in[3];
    const float* Ki  = (const float*)d_in[4];
    const float* Rr  = (const float*)d_in[5];
    const float* Ri  = (const float*)d_in[6];
    const float* br  = (const float*)d_in[7];
    const float* bi  = (const float*)d_in[8];
    float* out = (float*)d_out;

    unsigned short* Wt = (unsigned short*)d_ws;  // 1024*384*2 = 768 KB packed bf16 weights

    {
        int total = NTOT * KTOT;
        int blocks = (total + 255) / 256;
        pack_weights_kernel<<<blocks, 256, 0, stream>>>(Kr, Ki, Rr, Ri, Wt);
    }
    complex_lstm_kernel<<<BATCH / BROWS, 256, 0, stream>>>(x, h0, c0, Wt, br, bi, out);
}
